// Informer_77240691851863
// MI455X (gfx1250) — matrix operand order repaired and checked by reading, not compiled, not run on hardware
//
#include <hip/hip_runtime.h>
#include <hip/hip_bf16.h>

// ---------------------------------------------------------------------------
// Informer forward for MI455X (gfx1250, wave32, WMMA + TDM).
// Dense linears / attention GEMMs -> v_wmma_f32_16x16x32_bf16 (f32 accum),
// A-tiles streamed into LDS by the Tensor Data Mover (tensor_load_to_lds,
// TENSORcnt-paced, double-buffered). ProbSparse control flow is VALU.
// ---------------------------------------------------------------------------

#define DMODEL 512
#define NHEADS 8
#define HDIM   64
#define DFF    2048
#define BB     8
#define LENC   1024
#define LDEC   768
#define CIN    7
#define UTOP   21      // 3*ceil(ln(L)) == 21 for L in {512,768,1024}

#define BM 128
#define BN 128
#define BK 32

typedef __bf16 bf16_t;
typedef bf16_t bf16x16 __attribute__((ext_vector_type(16)));
typedef float  f32x8   __attribute__((ext_vector_type(8)));

union Frag { unsigned u[8]; bf16x16 v; };

#if defined(__has_builtin)
# if __has_builtin(__builtin_amdgcn_tensor_load_to_lds) && \
     __has_builtin(__builtin_amdgcn_s_wait_tensorcnt)
#  define HAVE_TDM 1
# endif
#endif
#ifndef HAVE_TDM
# define HAVE_TDM 0
#endif

__device__ __forceinline__ unsigned short f2bf(float f) {
  union { float f; unsigned u; } c; c.f = f;
  unsigned u = c.u;
  u += 0x7FFFu + ((u >> 16) & 1u);          // round-to-nearest-even
  return (unsigned short)(u >> 16);
}
__device__ __forceinline__ unsigned pack2(float a, float b) {
  return (unsigned)f2bf(a) | ((unsigned)f2bf(b) << 16);
}
__device__ __forceinline__ float gelu_f(float x) {
  return 0.5f * x * (1.f + erff(x * 0.70710678f));
}

#if HAVE_TDM
typedef unsigned u32x4v __attribute__((ext_vector_type(4)));
typedef int      i32x4v __attribute__((ext_vector_type(4)));
typedef int      i32x8v __attribute__((ext_vector_type(8)));

// TDM 2D tile load: rows x 32 fp32 tile (tile_dim0 = 32 cols, tile_dim1 = 128
// rows), tensor dims = remaining K/M so OOB rows/cols read back as zero.
// D# layout per CDNA5 ISA 8.3/8.4 (group0: count/lds/global/type;
// group1: data_size, dims, tile dims, dim0 stride).
__device__ __forceinline__ void tdm_load_tile(const float* g, void* l,
                                              int rows, int cols, int ldaE)
{
  unsigned long long ga = (unsigned long long)g;
  unsigned lofs = (unsigned)(unsigned long long)l;   // LDS byte offset (low 32b)
  u32x4v g0 = { 1u,                                   // count=1, user desc
                lofs,                                 // lds_addr
                (unsigned)ga,                         // global_addr[31:0]
                (unsigned)((ga >> 32) & 0x01FFFFFFull) | (2u << 30) }; // +type=2
  unsigned d0 = (unsigned)cols;   // tensor_dim0 (elements)
  unsigned d1 = (unsigned)rows;   // tensor_dim1 (rows)
  unsigned s0 = (unsigned)ldaE;   // tensor_dim0_stride (elements)
  i32x8v g1 = {
    (int)0x00020000u,                                        // data_size=2 (4B)
    (int)((d0 & 0xFFFFu) << 16),                             // dim0[15:0] @ [31:16]
    (int)(((d0 >> 16) & 0xFFFFu) | ((d1 & 0xFFFFu) << 16)),  // dim0 hi | dim1 lo
    (int)(((d1 >> 16) & 0xFFFFu) | (32u << 16)),             // dim1 hi | tile_dim0=32
    (int)(128u),                                             // tile_dim1=128
    (int)s0,                                                 // dim0 stride lo32
    0, 0
  };
  i32x4v gz = {0, 0, 0, 0};
#if __clang_major__ >= 23
  i32x8v gz8 = {0, 0, 0, 0, 0, 0, 0, 0};
  __builtin_amdgcn_tensor_load_to_lds(g0, g1, gz, gz, gz8, 0);
#else
  __builtin_amdgcn_tensor_load_to_lds(g0, g1, gz, gz, 0);
#endif
}
#endif  // HAVE_TDM

// ---------------------------------------------------------------------------
// Generic WMMA GEMM:  C[M,N] = act( alpha * A[M,K] x B + bias ) (+resid/+C)
//   bkm==0 : B given as W[N,K] row-major (computes A @ W^T)   (row stride ldb)
//   bkm==1 : B given as B[K,N] row-major                       (row stride ldb)
// Batched over blockIdx.z: z -> (zb = z/Hz, zh = z%Hz) with element strides.
// rowmap (optional): A row gather (circular-shifted rows for distil conv).
// Block: 256 thr = 8 waves; block tile 128x128; wave tile 32x64 (2x4 WMMA).
// A tile: TDM -> LDS fp32 double buffer (or sync fallback); bf16 convert at
// fragment build. B tile: sync load + bf16 convert into [k][n] LDS tile.
// ---------------------------------------------------------------------------
__global__ __launch_bounds__(256) void gemm_wmma(
    const float* __restrict__ A, long sAb, long sAh, int lda,
    const float* __restrict__ Bg, long sBb, long sBh, int ldb, int bkm,
    const float* __restrict__ bias,
    const float* __restrict__ resid,
    float* __restrict__ C, long sCb, long sCh, int ldc,
    int M, int N, int K, float alpha, int act, int accum, int Hz,
    const int* __restrict__ rowmap)
{
  __shared__ float          sA[2][BM][BK];     // fp32 A stage, double buffered
  __shared__ unsigned short lB[BK][BN + 4];    // [k][n] bf16 (padded)

  int z  = blockIdx.z;
  int zb = z / Hz, zh = z % Hz;
  A  += (long)zb * sAb + (long)zh * sAh;
  Bg += (long)zb * sBb + (long)zh * sBh;
  C  += (long)zb * sCb + (long)zh * sCh;
  const float* R = resid ? resid + (long)zb * sCb + (long)zh * sCh : nullptr;

  const int m0 = blockIdx.y * BM;
  const int n0 = blockIdx.x * BN;
  const int tid  = threadIdx.x;
  const int lane = tid & 31;
  const int w    = tid >> 5;
  const int wm   = (w & 3) * 32;   // wave row base in block tile
  const int wn   = (w >> 2) * 64;  // wave col base in block tile

  f32x8 acc[2][4];
#pragma unroll
  for (int mt = 0; mt < 2; ++mt)
#pragma unroll
    for (int nt = 0; nt < 4; ++nt)
      acc[mt][nt] = (f32x8){0.f, 0.f, 0.f, 0.f, 0.f, 0.f, 0.f, 0.f};

  const int lm = lane & 15;
  const int hi = lane >> 4;
  const int kg = hi * 8;           // A frag K group base
  const int bk = lm + hi * 16;     // B frag K row

  // cooperative sync A-tile stage (fallback + rowmap gather path)
  auto syncA = [&](int k0n, int buf) {
#pragma unroll
    for (int it = 0; it < 4; ++it) {
      int flat = tid + it * 256;               // 0..1023 float4 slots
      int r  = flat >> 3;                      // 0..127
      int c4 = (flat & 7) << 2;                // 0,4,..,28
      float4 f = {0.f, 0.f, 0.f, 0.f};
      int gm = m0 + r;
      if (gm < M) {
        int ar = rowmap ? rowmap[gm] : gm;
        f = *(const float4*)(A + (long)ar * lda + k0n + c4);
      }
      *(float4*)&sA[buf][r][c4] = f;
    }
  };

#if HAVE_TDM
  const bool tdm = (rowmap == nullptr);
  if (tdm) {
    if (tid < 32)
      tdm_load_tile(A + (long)m0 * lda, &sA[0][0][0], M - m0, K, lda);
  } else {
    syncA(0, 0);
  }
#else
  syncA(0, 0);
#endif

  const int nk = K / BK;
  for (int ik = 0; ik < nk; ++ik) {
    const int k0 = ik * BK;
    // ---- stage B tile (32 x 128) ---------------------------------------
    if (bkm == 0) {                // W[N][K] -> transpose into [k][n]
#pragma unroll
      for (int it = 0; it < 4; ++it) {
        int flat = tid + it * 256;
        int n  = flat >> 3;
        int c4 = (flat & 7) << 2;
        float4 f = {0.f, 0.f, 0.f, 0.f};
        int gn = n0 + n;
        if (gn < N) f = *(const float4*)(Bg + (long)gn * ldb + k0 + c4);
        lB[c4 + 0][n] = f2bf(f.x);
        lB[c4 + 1][n] = f2bf(f.y);
        lB[c4 + 2][n] = f2bf(f.z);
        lB[c4 + 3][n] = f2bf(f.w);
      }
    } else {                       // B[K][N] direct
#pragma unroll
      for (int it = 0; it < 4; ++it) {
        int flat = tid + it * 256;
        int kk = flat >> 5;                    // 0..31
        int c4 = (flat & 31) << 2;             // 0..124
        int gn = n0 + c4;
        const float* src = Bg + (long)(k0 + kk) * ldb + gn;
        float4 f = {0.f, 0.f, 0.f, 0.f};
        if (gn + 3 < N)      f = *(const float4*)src;
        else {
          if (gn + 0 < N) f.x = src[0];
          if (gn + 1 < N) f.y = src[1];
          if (gn + 2 < N) f.z = src[2];
        }
        *(unsigned*)&lB[kk][c4]     = pack2(f.x, f.y);
        *(unsigned*)&lB[kk][c4 + 2] = pack2(f.z, f.w);
      }
    }

    // ---- A tile for this step / prefetch next --------------------------
#if HAVE_TDM
    if (tdm) {
      if (tid < 32) {
        if (ik + 1 < nk)
          tdm_load_tile(A + (long)m0 * lda + (k0 + BK),
                        &sA[(ik + 1) & 1][0][0], M - m0, K - (k0 + BK), lda);
        if (ik + 1 < nk) __builtin_amdgcn_s_wait_tensorcnt(1);
        else             __builtin_amdgcn_s_wait_tensorcnt(0);
      }
    } else {
      if (ik > 0) syncA(k0, ik & 1);
    }
#else
    if (ik > 0) syncA(k0, ik & 1);
#endif
    __syncthreads();   // tile (A fp32 + B bf16) visible to all waves

    // ---- WMMA math ------------------------------------------------------
    const float (*As)[BK] = sA[ik & 1];
    Frag af[2];
#pragma unroll
    for (int mt = 0; mt < 2; ++mt) {
      int row = wm + mt * 16 + lm;
#pragma unroll
      for (int j = 0; j < 4; ++j) {
        af[mt].u[j]     = pack2(As[row][kg + 2 * j],      As[row][kg + 2 * j + 1]);
        af[mt].u[4 + j] = pack2(As[row][16 + kg + 2 * j], As[row][16 + kg + 2 * j + 1]);
      }
    }
#pragma unroll
    for (int nt = 0; nt < 4; ++nt) {
      Frag bf;
      int col = wn + nt * 16;
#pragma unroll
      for (int j = 0; j < 8; ++j) {
        int kb = (j < 4) ? (kg + 2 * j) : (16 + kg + 2 * (j - 4));
        bf.u[j] = (unsigned)lB[kb][col + lm] | ((unsigned)lB[kb + 1][col + lm] << 16);
      }
#pragma unroll
      for (int mt = 0; mt < 2; ++mt)
        acc[mt][nt] = __builtin_amdgcn_wmma_f32_16x16x32_bf16(
            false, af[mt].v, false, bf.v, (short)0, acc[mt][nt], false, false);
    }
    __syncthreads();   // safe to overwrite LDS next iteration
  }

  // ---- epilogue: C layout = VGPR r: (m = hi*8+r, n = lm) -----------------
#pragma unroll
  for (int mt = 0; mt < 2; ++mt) {
#pragma unroll
    for (int nt = 0; nt < 4; ++nt) {
      int col = n0 + wn + nt * 16 + lm;
      if (col >= N) continue;
      float bv = bias ? bias[col] : 0.f;
#pragma unroll
      for (int r = 0; r < 8; ++r) {
        int row = m0 + wm + mt * 16 + hi * 8 + r;
        if (row >= M) continue;
        float v = acc[mt][nt][r] * alpha + bv;
        if (act == 1) v = gelu_f(v);
        long off = (long)row * ldc + col;
        if (R)     v += R[off];
        if (accum) v += C[off];
        C[off] = v;
      }
    }
  }
}

// ---------------------------------------------------------------------------
// Embedding: circular conv1d (Cin->512, k=3) + sinusoidal pos + temporal emb
// ---------------------------------------------------------------------------
__global__ __launch_bounds__(128) void embed_k(
    const float* __restrict__ x, const int* __restrict__ mark,
    const float* __restrict__ w, int L, float* __restrict__ out)
{
  int bl = blockIdx.x;
  int b = bl / L, l = bl % L;
  __shared__ float xs[3][CIN];
  __shared__ int   ms[5];
  int tid = threadIdx.x;
  if (tid < 3 * CIN) {
    int t = tid / CIN, i = tid % CIN;
    int li = (l + t - 1 + L) % L;
    xs[t][i] = x[((long)b * L + li) * CIN + i];
  }
  if (tid < 5) ms[tid] = mark[((long)b * L + l) * 5 + tid];
  __syncthreads();
  const float c0 = -9.21034037f / (float)DMODEL;   // -ln(1e4)/512
#pragma unroll
  for (int oo = 0; oo < 4; ++oo) {
    int o = tid + oo * 128;
    float acc = 0.f;
#pragma unroll
    for (int t = 0; t < 3; ++t)
#pragma unroll
      for (int i = 0; i < CIN; ++i)
        acc += xs[t][i] * w[(o * CIN + i) * 3 + t];
    float f = __expf((float)(o & ~1) * c0);
    bool even = !(o & 1);
    float a = (float)l * f;
    acc += even ? __sinf(a) : __cosf(a);
#pragma unroll
    for (int i = 0; i < 5; ++i) {
      float am = (float)ms[i] * f;
      acc += even ? __sinf(am) : __cosf(am);
    }
    out[((long)b * L + l) * DMODEL + o] = acc;
  }
}

// ---------------------------------------------------------------------------
// LayerNorm over 512: out = LN(x [+ res]) * g + b ; one wave per row
// ---------------------------------------------------------------------------
__global__ __launch_bounds__(256) void layernorm_k(
    const float* __restrict__ x, const float* __restrict__ res,
    const float* __restrict__ g, const float* __restrict__ bb,
    float* __restrict__ out, int M)
{
  int row  = blockIdx.x * 8 + (threadIdx.x >> 5);
  int lane = threadIdx.x & 31;
  if (row >= M) return;
  const float4* xp = (const float4*)(x + (long)row * DMODEL);
  const float4* rp = res ? (const float4*)(res + (long)row * DMODEL) : nullptr;
  float4 v[4];
  float s = 0.f, s2 = 0.f;
#pragma unroll
  for (int i = 0; i < 4; ++i) {
    float4 t = xp[lane + 32 * i];
    if (rp) {
      float4 r2 = rp[lane + 32 * i];
      t.x += r2.x; t.y += r2.y; t.z += r2.z; t.w += r2.w;
    }
    v[i] = t;
    s  += t.x + t.y + t.z + t.w;
    s2 += t.x * t.x + t.y * t.y + t.z * t.z + t.w * t.w;
  }
#pragma unroll
  for (int m = 16; m > 0; m >>= 1) {
    s  += __shfl_xor(s,  m, 32);
    s2 += __shfl_xor(s2, m, 32);
  }
  float mean = s / 512.f;
  float var  = s2 / 512.f - mean * mean;
  float inv  = rsqrtf(var + 1e-5f);
  float4* op = (float4*)(out + (long)row * DMODEL);
#pragma unroll
  for (int i = 0; i < 4; ++i) {
    float4 gg = ((const float4*)g)[lane + 32 * i];
    float4 bv = ((const float4*)bb)[lane + 32 * i];
    float4 o;
    o.x = (v[i].x - mean) * inv * gg.x + bv.x;
    o.y = (v[i].y - mean) * inv * gg.y + bv.y;
    o.z = (v[i].z - mean) * inv * gg.z + bv.z;
    o.w = (v[i].w - mean) * inv * gg.w + bv.w;
    op[lane + 32 * i] = o;
  }
}

// Softmax over last dim (width multiple of 128, <=1024); one wave per row
__global__ __launch_bounds__(256) void softmax_k(float* __restrict__ s, int rows, int width)
{
  int row  = blockIdx.x * 8 + (threadIdx.x >> 5);
  int lane = threadIdx.x & 31;
  if (row >= rows) return;
  float4* p = (float4*)(s + (long)row * width);
  int nf = width >> 7;
  float4 v[8];
  float mx = -1e30f;
  for (int i = 0; i < nf; ++i) {
    v[i] = p[lane + 32 * i];
    mx = fmaxf(mx, fmaxf(fmaxf(v[i].x, v[i].y), fmaxf(v[i].z, v[i].w)));
  }
#pragma unroll
  for (int m = 16; m > 0; m >>= 1) mx = fmaxf(mx, __shfl_xor(mx, m, 32));
  float sum = 0.f;
  for (int i = 0; i < nf; ++i) {
    v[i].x = __expf(v[i].x - mx); v[i].y = __expf(v[i].y - mx);
    v[i].z = __expf(v[i].z - mx); v[i].w = __expf(v[i].w - mx);
    sum += v[i].x + v[i].y + v[i].z + v[i].w;
  }
#pragma unroll
  for (int m = 16; m > 0; m >>= 1) sum += __shfl_xor(sum, m, 32);
  float inv = 1.f / sum;
  for (int i = 0; i < nf; ++i) {
    v[i].x *= inv; v[i].y *= inv; v[i].z *= inv; v[i].w *= inv;
    p[lane + 32 * i] = v[i];
  }
}

// ---------------------------------------------------------------------------
// ProbSparse attention helpers
// ---------------------------------------------------------------------------
__device__ __forceinline__ unsigned hash_u(unsigned x) {
  x *= 2654435761u; x ^= x >> 16; x *= 2246822519u; x ^= x >> 13;
  return x;
}

// M[b,h,l] = max_j(q.k_samp_j) - sum_j/Lk ; one wave per (b,h,l)
__global__ __launch_bounds__(256) void prob_measure_k(
    const float* __restrict__ q, const float* __restrict__ k,
    float* __restrict__ Mout, int L, int Lk, unsigned seed)
{
  int wid  = blockIdx.x * 8 + (threadIdx.x >> 5);
  int lane = threadIdx.x & 31;
  int tot  = BB * NHEADS * L;
  if (wid >= tot) return;
  int l  = wid % L;
  int bh = wid / L;
  int h  = bh % NHEADS, b = bh / NHEADS;
  const float* qp = q + ((long)b * L + l) * DMODEL + h * HDIM;
  bool act = lane < UTOP;
  float dot = 0.f;
  if (act) {
    int ki = (int)(hash_u((unsigned)(l * UTOP + lane) + seed * 747796405u) % (unsigned)Lk);
    const float* kp = k + ((long)b * Lk + ki) * DMODEL + h * HDIM;
#pragma unroll 8
    for (int d = 0; d < HDIM; ++d) dot += qp[d] * kp[d];
  }
  float mx = act ? dot : -1e30f;
  float sm = act ? dot : 0.f;
#pragma unroll
  for (int m = 16; m > 0; m >>= 1) {
    mx = fmaxf(mx, __shfl_xor(mx, m, 32));
    sm += __shfl_xor(sm, m, 32);
  }
  if (lane == 0) Mout[wid] = mx - sm / (float)Lk;
}

// top-21 indices of M[L] per (b,h); one block per (b,h)
__global__ __launch_bounds__(256) void topk_k(
    const float* __restrict__ Mm, int L, int* __restrict__ top)
{
  __shared__ float sv[1024];
  __shared__ float rv[256];
  __shared__ int   ri[256];
  int bh = blockIdx.x, tid = threadIdx.x;
  const float* row = Mm + (long)bh * L;
  for (int i = tid; i < L; i += 256) sv[i] = row[i];
  __syncthreads();
  for (int it = 0; it < UTOP; ++it) {
    float best = -1e30f; int bi = 0;
    for (int i = tid; i < L; i += 256)
      if (sv[i] > best) { best = sv[i]; bi = i; }
    rv[tid] = best; ri[tid] = bi;
    __syncthreads();
    for (int s = 128; s > 0; s >>= 1) {
      if (tid < s && rv[tid + s] > rv[tid]) { rv[tid] = rv[tid + s]; ri[tid] = ri[tid + s]; }
      __syncthreads();
    }
    if (tid == 0) { top[bh * UTOP + it] = ri[0]; sv[ri[0]] = -1e30f; }
    __syncthreads();
  }
}

// encoder ctx init: broadcast mean_s(V); ctx layout [b][l][h*64+d]
__global__ void ctx_mean_k(const float* __restrict__ v, float* __restrict__ ctx, int L)
{
  int i = blockIdx.x * 256 + threadIdx.x;
  if (i >= BB * NHEADS * HDIM) return;
  int d = i % HDIM, h = (i / HDIM) % NHEADS, b = i / (HDIM * NHEADS);
  float s = 0.f;
  for (int l = 0; l < L; ++l) s += v[((long)b * L + l) * DMODEL + h * HDIM + d];
  s /= (float)L;
  for (int l = 0; l < L; ++l) ctx[((long)b * L + l) * DMODEL + h * HDIM + d] = s;
}

// decoder self ctx init: cumsum_s(V); mixed layout [b][h][l][d]
__global__ void ctx_cumsum_k(const float* __restrict__ v, float* __restrict__ ctx, int L)
{
  int i = blockIdx.x * 256 + threadIdx.x;
  if (i >= BB * NHEADS * HDIM) return;
  int d = i % HDIM, h = (i / HDIM) % NHEADS, b = i / (HDIM * NHEADS);
  float s = 0.f;
  for (int l = 0; l < L; ++l) {
    s += v[((long)b * L + l) * DMODEL + h * HDIM + d];
    ctx[(((long)b * NHEADS + h) * L + l) * HDIM + d] = s;
  }
}

// softmax(q_sel . K / 8) @ V for each selected query; one block per (b,h,u)
__global__ __launch_bounds__(256) void prob_update_k(
    const float* __restrict__ q, const float* __restrict__ k,
    const float* __restrict__ v, const int* __restrict__ top,
    float* __restrict__ ctx, int Lq, int Lk, int mixed, int causal)
{
  __shared__ float sc[1024];
  __shared__ float qv[HDIM];
  __shared__ float red[256];
  __shared__ float part[4][HDIM];
  int blk = blockIdx.x;
  int ui = blk % UTOP;
  int bh = blk / UTOP;
  int h = bh % NHEADS, b = bh / NHEADS;
  int tid = threadIdx.x;
  int l0 = top[bh * UTOP + ui];
  if (tid < HDIM) qv[tid] = q[((long)b * Lq + l0) * DMODEL + h * HDIM + tid];
  __syncthreads();
  float lmax = -1e30f;
  for (int kk = tid; kk < Lk; kk += 256) {
    const float* kp = k + ((long)b * Lk + kk) * DMODEL + h * HDIM;
    float d = 0.f;
#pragma unroll 8
    for (int j = 0; j < HDIM; ++j) d += qv[j] * kp[j];
    d *= 0.125f;
    if (causal && kk > l0) d = -1e9f;
    sc[kk] = d;
    lmax = fmaxf(lmax, d);
  }
  red[tid] = lmax;
  __syncthreads();
  for (int s = 128; s > 0; s >>= 1) {
    if (tid < s) red[tid] = fmaxf(red[tid], red[tid + s]);
    __syncthreads();
  }
  float mx = red[0];
  __syncthreads();
  float lsum = 0.f;
  for (int kk = tid; kk < Lk; kk += 256) {
    float e = __expf(sc[kk] - mx);
    sc[kk] = e;
    lsum += e;
  }
  red[tid] = lsum;
  __syncthreads();
  for (int s = 128; s > 0; s >>= 1) {
    if (tid < s) red[tid] += red[tid + s];
    __syncthreads();
  }
  float inv = 1.f / red[0];
  __syncthreads();
  int d  = tid & 63;
  int ch = tid >> 6;
  float acc = 0.f;
  for (int kk = ch; kk < Lk; kk += 4)
    acc += sc[kk] * v[((long)b * Lk + kk) * DMODEL + h * HDIM + d];
  part[ch][d] = acc;
  __syncthreads();
  if (tid < HDIM) {
    float o = (part[0][tid] + part[1][tid] + part[2][tid] + part[3][tid]) * inv;
    long off = mixed ? ((((long)b * NHEADS + h) * Lq + l0) * HDIM + tid)
                     : (((long)b * Lq + l0) * DMODEL + h * HDIM + tid);
    ctx[off] = o;
  }
}

// ---------------------------------------------------------------------------
// Distil helpers
// ---------------------------------------------------------------------------
__global__ void rowmap_k(int* __restrict__ maps, int L)
{
  int i = blockIdx.x * 256 + threadIdx.x;
  int tot = BB * L;
  if (i >= 3 * tot) return;
  int t = i / tot, r = i % tot;
  int b = r / L, l = r % L;
  maps[t * tot + r] = b * L + ((l + t - 1 + L) % L);
}
__global__ void wrepack_k(const float* __restrict__ w, float* __restrict__ wt)
{
  int i = blockIdx.x * 256 + threadIdx.x;
  if (i >= 3 * DMODEL * DMODEL) return;
  int t = i / (DMODEL * DMODEL), oi = i % (DMODEL * DMODEL);
  wt[t * DMODEL * DMODEL + oi] = w[oi * 3 + t];
}
// BN (rsqrt(1+eps) affine) + ELU + maxpool(k=3,s=2,pad=1)
__global__ void distil_post_k(const float* __restrict__ y,
                              const float* __restrict__ g, const float* __restrict__ bb,
                              float* __restrict__ out, int L)
{
  int i = blockIdx.x * 256 + threadIdx.x;
  int Lo = L / 2;
  if (i >= BB * Lo * DMODEL) return;
  int c = i % DMODEL;
  int r = i / DMODEL;
  int lo = r % Lo, b = r / Lo;
  const float rs = rsqrtf(1.f + 1e-5f);
  float m = -1e30f;
  for (int dl = -1; dl <= 1; ++dl) {
    int li = 2 * lo + dl;
    if (li < 0 || li >= L) continue;
    float vv = y[((long)b * L + li) * DMODEL + c] * rs * g[c] + bb[c];
    vv = vv > 0.f ? vv : (__expf(vv) - 1.f);
    m = fmaxf(m, vv);
  }
  out[((long)b * Lo + lo) * DMODEL + c] = m;
}

// final projection: out[b, l, c] = dec[b, 512+l, :] . W[c,:] + bias[c]
__global__ void proj_k(const float* __restrict__ dec, const float* __restrict__ W,
                       const float* __restrict__ bb, float* __restrict__ out)
{
  int i = blockIdx.x * 256 + threadIdx.x;
  if (i >= BB * 256 * 7) return;
  int c = i % 7;
  int l = (i / 7) % 256;
  int b = i / (7 * 256);
  const float* xr = dec + ((long)b * LDEC + (LDEC - 256 + l)) * DMODEL;
  float s = bb[c];
#pragma unroll 8
  for (int j = 0; j < DMODEL; ++j) s += xr[j] * W[c * DMODEL + j];
  out[i] = s;
}

// ---------------------------------------------------------------------------
static void gemm(hipStream_t st,
                 const float* A, long sAb, long sAh, int lda,
                 const float* B_, long sBb, long sBh, int ldb, int bkm,
                 const float* bias, const float* resid,
                 float* C, long sCb, long sCh, int ldc,
                 int M, int N, int K, float alpha, int act, int accum,
                 int Hz, int Z, const int* rowmap)
{
  dim3 grid((N + BN - 1) / BN, (M + BM - 1) / BM, Z);
  gemm_wmma<<<grid, 256, 0, st>>>(A, sAb, sAh, lda, B_, sBb, sBh, ldb, bkm,
                                  bias, resid, C, sCb, sCh, ldc,
                                  M, N, K, alpha, act, accum, Hz, rowmap);
}

// Parameter leaf indices (jax pytree: dict keys sorted, lists in order):
//  0 x_enc  1 x_dec
//  conv_layers[0]: 2 b, 3 bn_b, 4 bn_g, 5 w
//  6 dec_conv
//  dec_layers[0]: conv1{7 W,8 b} conv2{9 W,10 b}
//    cross{k:11W,12b o:13W,14b q:15W,16b v:17W,18b}
//    norm1{19 b,20 g} norm2{21 b,22 g} norm3{23 b,24 g}
//    self{k:25W,26b o:27W,28b q:29W,30b v:31W,32b}
//  dec_norm{33 b,34 g}
//  35 enc_conv
//  enc_layers[i] base 36 / 52: k{+0W,+1b} o{+2,+3} q{+4,+5} v{+6,+7}
//    conv1{+8,+9} conv2{+10,+11} norm1{+12b,+13g} norm2{+14b,+15g}
//  enc_norm{68 b,69 g}   proj{70 W,71 b}
//  72 x_mark_enc (int)   73 x_mark_dec (int)
extern "C" void kernel_launch(void* const* d_in, const int* in_sizes, int n_in,
                              void* d_out, int out_size, void* d_ws, size_t ws_size,
                              hipStream_t stream)
{
  (void)in_sizes; (void)n_in; (void)out_size; (void)ws_size;
  auto F = [&](int i) { return (const float*)d_in[i]; };
  const int* mark_enc = (const int*)d_in[72];
  const int* mark_dec = (const int*)d_in[73];

  char* ws = (char*)d_ws;
  size_t off = 0;
  auto alloc = [&](size_t bytes) {
    size_t r = off;
    off += (bytes + 255) & ~(size_t)255;
    return r;
  };
  float* X   = (float*)(ws + alloc((size_t)BB * LENC * DMODEL * 4));   // enc stream
  float* Y   = (float*)(ws + alloc((size_t)BB * LENC * DMODEL * 4));   // temp
  float* Qb  = (float*)(ws + alloc((size_t)BB * LENC * DMODEL * 4));
  float* Kb  = (float*)(ws + alloc((size_t)BB * LENC * DMODEL * 4));
  float* Vb  = (float*)(ws + alloc((size_t)BB * LENC * DMODEL * 4));
  float* Hh  = (float*)(ws + alloc((size_t)BB * LENC * DFF * 4));      // FFN hidden (aliased as CTX)
  float* DX  = (float*)(ws + alloc((size_t)BB * LDEC * DMODEL * 4));   // dec stream
  float* DY  = (float*)(ws + alloc((size_t)BB * LDEC * DMODEL * 4));
  float* SC  = (float*)(ws + alloc((size_t)NHEADS * LDEC * 512 * 4));  // per-b cross scores
  float* Mm  = (float*)(ws + alloc((size_t)BB * NHEADS * LENC * 4));
  int*   TOP = (int*)  (ws + alloc((size_t)BB * NHEADS * UTOP * 4));
  int*   RM  = (int*)  (ws + alloc((size_t)3 * BB * LENC * 4));
  float* WT  = (float*)(ws + alloc((size_t)3 * DMODEL * DMODEL * 4));
  float* CTX = Hh;   // attention-context buffer, time-shares with FFN hidden

  // ===== Encoder ===========================================================
  embed_k<<<BB * LENC, 128, 0, stream>>>(F(0), mark_enc, F(35), LENC, X);

  const int Ls[2] = {LENC, LENC / 2};
  const int baseIdx[2] = {36, 52};
  for (int li = 0; li < 2; ++li) {
    int L = Ls[li];
    int Mr = BB * L;
    int bi = baseIdx[li];
    const float *Wk = F(bi + 0), *bk = F(bi + 1), *Wo = F(bi + 2), *bo = F(bi + 3);
    const float *Wq = F(bi + 4), *bq = F(bi + 5), *Wv = F(bi + 6), *bv = F(bi + 7);
    const float *W1 = F(bi + 8), *b1 = F(bi + 9), *W2 = F(bi + 10), *b2 = F(bi + 11);
    const float *n1b = F(bi + 12), *n1g = F(bi + 13), *n2b = F(bi + 14), *n2g = F(bi + 15);

    gemm(stream, X, 0, 0, DMODEL, Wq, 0, 0, DMODEL, 0, bq, nullptr,
         Qb, 0, 0, DMODEL, Mr, DMODEL, DMODEL, 1.f, 0, 0, 1, 1, nullptr);
    gemm(stream, X, 0, 0, DMODEL, Wk, 0, 0, DMODEL, 0, bk, nullptr,
         Kb, 0, 0, DMODEL, Mr, DMODEL, DMODEL, 1.f, 0, 0, 1, 1, nullptr);
    gemm(stream, X, 0, 0, DMODEL, Wv, 0, 0, DMODEL, 0, bv, nullptr,
         Vb, 0, 0, DMODEL, Mr, DMODEL, DMODEL, 1.f, 0, 0, 1, 1, nullptr);

    int waves = BB * NHEADS * L;
    prob_measure_k<<<(waves + 7) / 8, 256, 0, stream>>>(Qb, Kb, Mm, L, L, (unsigned)li);
    topk_k<<<BB * NHEADS, 256, 0, stream>>>(Mm, L, TOP);
    ctx_mean_k<<<(BB * NHEADS * HDIM + 255) / 256, 256, 0, stream>>>(Vb, CTX, L);
    prob_update_k<<<BB * NHEADS * UTOP, 256, 0, stream>>>(Qb, Kb, Vb, TOP, CTX, L, L, 0, 0);

    gemm(stream, CTX, 0, 0, DMODEL, Wo, 0, 0, DMODEL, 0, bo, nullptr,
         Y, 0, 0, DMODEL, Mr, DMODEL, DMODEL, 1.f, 0, 0, 1, 1, nullptr);
    layernorm_k<<<(Mr + 7) / 8, 256, 0, stream>>>(X, Y, n1g, n1b, X, Mr);

    gemm(stream, X, 0, 0, DMODEL, W1, 0, 0, DMODEL, 0, b1, nullptr,
         Hh, 0, 0, DFF, Mr, DFF, DMODEL, 1.f, 1, 0, 1, 1, nullptr);   // GELU fused
    gemm(stream, Hh, 0, 0, DFF, W2, 0, 0, DFF, 0, b2, nullptr,
         Y, 0, 0, DMODEL, Mr, DMODEL, DFF, 1.f, 0, 0, 1, 1, nullptr);
    layernorm_k<<<(Mr + 7) / 8, 256, 0, stream>>>(X, Y, n2g, n2b, X, Mr);

    if (li == 0) {  // distil: circular conv(k=3) via 3 gathered GEMM taps
      rowmap_k<<<(3 * BB * LENC + 255) / 256, 256, 0, stream>>>(RM, LENC);
      wrepack_k<<<(3 * DMODEL * DMODEL + 255) / 256, 256, 0, stream>>>(F(5), WT);
      for (int t = 0; t < 3; ++t)
        gemm(stream, X, 0, 0, DMODEL, WT + (size_t)t * DMODEL * DMODEL, 0, 0, DMODEL, 0,
             (t == 0 ? F(2) : nullptr), nullptr, Hh, 0, 0, DMODEL,
             Mr, DMODEL, DMODEL, 1.f, 0, (t > 0), 1, 1, RM + t * BB * LENC);
      distil_post_k<<<(BB * (LENC / 2) * DMODEL + 255) / 256, 256, 0, stream>>>(
          Hh, F(4), F(3), X, LENC);
    }
  }
  layernorm_k<<<(BB * 512 + 7) / 8, 256, 0, stream>>>(X, nullptr, F(69), F(68), X, BB * 512);

  // ===== Decoder ===========================================================
  embed_k<<<BB * LDEC, 128, 0, stream>>>(F(1), mark_dec, F(6), LDEC, DX);
  {
    const int L = LDEC, Mr = BB * LDEC;
    const int Senc = 512, Mk = BB * Senc;
    const float *sWk = F(25), *sbk = F(26), *sWo = F(27), *sbo = F(28);
    const float *sWq = F(29), *sbq = F(30), *sWv = F(31), *sbv = F(32);
    const float *cWk = F(11), *cbk = F(12), *cWo = F(13), *cbo = F(14);
    const float *cWq = F(15), *cbq = F(16), *cWv = F(17), *cbv = F(18);
    const float *W1 = F(7), *b1 = F(8), *W2 = F(9), *b2 = F(10);

    // --- self ProbSparse (causal, mixed output layout) ---
    gemm(stream, DX, 0, 0, DMODEL, sWq, 0, 0, DMODEL, 0, sbq, nullptr,
         Qb, 0, 0, DMODEL, Mr, DMODEL, DMODEL, 1.f, 0, 0, 1, 1, nullptr);
    gemm(stream, DX, 0, 0, DMODEL, sWk, 0, 0, DMODEL, 0, sbk, nullptr,
         Kb, 0, 0, DMODEL, Mr, DMODEL, DMODEL, 1.f, 0, 0, 1, 1, nullptr);
    gemm(stream, DX, 0, 0, DMODEL, sWv, 0, 0, DMODEL, 0, sbv, nullptr,
         Vb, 0, 0, DMODEL, Mr, DMODEL, DMODEL, 1.f, 0, 0, 1, 1, nullptr);
    int waves = BB * NHEADS * L;
    prob_measure_k<<<(waves + 7) / 8, 256, 0, stream>>>(Qb, Kb, Mm, L, L, 100u);
    topk_k<<<BB * NHEADS, 256, 0, stream>>>(Mm, L, TOP);
    ctx_cumsum_k<<<(BB * NHEADS * HDIM + 255) / 256, 256, 0, stream>>>(Vb, CTX, L);
    prob_update_k<<<BB * NHEADS * UTOP, 256, 0, stream>>>(Qb, Kb, Vb, TOP, CTX, L, L, 1, 1);
    gemm(stream, CTX, 0, 0, DMODEL, sWo, 0, 0, DMODEL, 0, sbo, nullptr,
         DY, 0, 0, DMODEL, Mr, DMODEL, DMODEL, 1.f, 0, 0, 1, 1, nullptr);
    layernorm_k<<<(Mr + 7) / 8, 256, 0, stream>>>(DX, DY, F(20), F(19), DX, Mr);

    // --- cross full attention against encoder output (S=512) ---
    gemm(stream, DX, 0, 0, DMODEL, cWq, 0, 0, DMODEL, 0, cbq, nullptr,
         Qb, 0, 0, DMODEL, Mr, DMODEL, DMODEL, 1.f, 0, 0, 1, 1, nullptr);
    gemm(stream, X, 0, 0, DMODEL, cWk, 0, 0, DMODEL, 0, cbk, nullptr,
         Kb, 0, 0, DMODEL, Mk, DMODEL, DMODEL, 1.f, 0, 0, 1, 1, nullptr);
    gemm(stream, X, 0, 0, DMODEL, cWv, 0, 0, DMODEL, 0, cbv, nullptr,
         Vb, 0, 0, DMODEL, Mk, DMODEL, DMODEL, 1.f, 0, 0, 1, 1, nullptr);
    for (int b = 0; b < BB; ++b) {
      // scores[h, l, s] = Q . K^T / 8   (batched over z = h)
      gemm(stream, Qb + (long)b * LDEC * DMODEL, 0, HDIM, DMODEL,
           Kb + (long)b * Senc * DMODEL, 0, HDIM, DMODEL, 0,
           nullptr, nullptr, SC, 0, (long)LDEC * Senc, Senc,
           LDEC, Senc, HDIM, 0.125f, 0, 0, NHEADS, NHEADS, nullptr);
      softmax_k<<<(NHEADS * LDEC + 7) / 8, 256, 0, stream>>>(SC, NHEADS * LDEC, Senc);
      // ctx[l, h*64+d] = P @ V          (B k-major layout)
      gemm(stream, SC, 0, (long)LDEC * Senc, Senc,
           Vb + (long)b * Senc * DMODEL, 0, HDIM, DMODEL, 1,
           nullptr, nullptr, CTX + (long)b * LDEC * DMODEL, 0, HDIM, DMODEL,
           LDEC, HDIM, Senc, 1.f, 0, 0, NHEADS, NHEADS, nullptr);
    }
    gemm(stream, CTX, 0, 0, DMODEL, cWo, 0, 0, DMODEL, 0, cbo, nullptr,
         DY, 0, 0, DMODEL, Mr, DMODEL, DMODEL, 1.f, 0, 0, 1, 1, nullptr);
    layernorm_k<<<(Mr + 7) / 8, 256, 0, stream>>>(DX, DY, F(22), F(21), DX, Mr);

    // --- FFN ---
    gemm(stream, DX, 0, 0, DMODEL, W1, 0, 0, DMODEL, 0, b1, nullptr,
         Hh, 0, 0, DFF, Mr, DFF, DMODEL, 1.f, 1, 0, 1, 1, nullptr);
    gemm(stream, Hh, 0, 0, DFF, W2, 0, 0, DFF, 0, b2, nullptr,
         DY, 0, 0, DMODEL, Mr, DMODEL, DFF, 1.f, 0, 0, 1, 1, nullptr);
    layernorm_k<<<(Mr + 7) / 8, 256, 0, stream>>>(DX, DY, F(24), F(23), DX, Mr);
  }
  layernorm_k<<<(BB * LDEC + 7) / 8, 256, 0, stream>>>(DX, nullptr, F(34), F(33), DX, BB * LDEC);
  proj_k<<<(BB * 256 * 7 + 255) / 256, 256, 0, stream>>>(DX, F(70), F(71), (float*)d_out);
}
